// StandardMultiHeadAttention_62972810494169
// MI455X (gfx1250) — compile-verified, matrix operand order checked by reading, not run on hardware
//
#include <hip/hip_runtime.h>
#include <hip/hip_bf16.h>

typedef __bf16 bf16;
typedef __attribute__((ext_vector_type(16))) __bf16 v16bf;
typedef __attribute__((ext_vector_type(8)))  __bf16 v8bf;
typedef __attribute__((ext_vector_type(8)))  float  v8f;

#define DM     1024
#define HEADS  16
#define HD     64
#define SEQ    2048
#define BATCH  4
#define ROWS   (BATCH*SEQ)   // 8192

// ---------------------------------------------------------------------------
// WMMA fragment loader (bf16, 16x16x32).
// Per ISA 7.12.2: for A (16x32) lane L holds row M=L%16; lane-half h=L/16
// holds K in two contiguous runs of 8: [8h,8h+8) and [16+8h,16+8h+8).
// We assume the symmetric layout for B staged N-major (row n holds K contig),
// so one loader serves both operands: two 16-byte LDS loads per fragment.
// ---------------------------------------------------------------------------
static __device__ __forceinline__ v16bf frag_load(const bf16* rowptr, int h) {
  const v8bf lo = *(const v8bf*)(rowptr + 8 * h);
  const v8bf hi = *(const v8bf*)(rowptr + 16 + 8 * h);
  v16bf r;
#pragma unroll
  for (int i = 0; i < 8; ++i) { r[i] = lo[i]; r[8 + i] = hi[i]; }
  return r;
}

static __device__ __forceinline__ v8f wmma_bf16(v16bf a, v16bf b, v8f c) {
  return __builtin_amdgcn_wmma_f32_16x16x32_bf16(false, a, false, b,
                                                 (short)0, c, false, false);
}

// ---------------------------------------------------------------------------
// fp32 -> bf16 conversion (vector x4)
// ---------------------------------------------------------------------------
__global__ void cvt_f32_bf16(const float* __restrict__ src,
                             bf16* __restrict__ dst, int n) {
  int i = (blockIdx.x * 256 + threadIdx.x) * 4;
  if (i < n) {
    float4 f = *(const float4*)(src + i);
    dst[i + 0] = (bf16)f.x;
    dst[i + 1] = (bf16)f.y;
    dst[i + 2] = (bf16)f.z;
    dst[i + 3] = (bf16)f.w;
  }
}

// ---------------------------------------------------------------------------
// QKV projection: Y = X @ W^T + b, M=8192, N=1024, K=1024, bf16 WMMA.
// blockIdx.z selects {Q,K,V}. Q,K stored [b,h,t,d]; V stored transposed
// [b,h,d,t] so the PV GEMM B-operand reads contiguous K(t).
// Block tile 128x128x32; 8 waves, each wave 2x4 16x16 tiles.
// ---------------------------------------------------------------------------
#define BK   32
#define LDAB 48   // 32 + 16 pad halves; 96B rows (16B aligned)

__global__ __launch_bounds__(256)
void gemm_qkv(const bf16* __restrict__ X,
              const bf16* __restrict__ Wqb, const bf16* __restrict__ Wkb,
              const bf16* __restrict__ Wvb,
              const float* __restrict__ bq, const float* __restrict__ bk,
              const float* __restrict__ bv,
              bf16* __restrict__ Qo, bf16* __restrict__ Ko,
              bf16* __restrict__ Vto) {
  const int proj = blockIdx.z;
  const bf16*  W    = (proj == 0) ? Wqb : (proj == 1) ? Wkb : Wvb;
  const float* bias = (proj == 0) ? bq  : (proj == 1) ? bk  : bv;

  const int m0 = blockIdx.y * 128;
  const int n0 = blockIdx.x * 128;
  const int tid = threadIdx.x;
  const int lane = tid & 31, wave = tid >> 5;
  const int l15 = lane & 15, hl = lane >> 4;
  const int wm = wave & 3, wn = wave >> 2;

  __shared__ __align__(16) bf16 As[128 * LDAB];
  __shared__ __align__(16) bf16 Bs[128 * LDAB];

  v8f acc[2][4];
#pragma unroll
  for (int mt = 0; mt < 2; ++mt)
#pragma unroll
    for (int nt = 0; nt < 4; ++nt)
#pragma unroll
      for (int r = 0; r < 8; ++r) acc[mt][nt][r] = 0.0f;

  const int row = tid >> 1, seg = tid & 1;

  for (int k0 = 0; k0 < DM; k0 += BK) {
    __syncthreads();
    {
      const bf16* srcA = X + (size_t)(m0 + row) * DM + k0 + seg * 16;
      const bf16* srcB = W + (size_t)(n0 + row) * DM + k0 + seg * 16;
      v8bf a0 = ((const v8bf*)srcA)[0], a1 = ((const v8bf*)srcA)[1];
      v8bf b0 = ((const v8bf*)srcB)[0], b1 = ((const v8bf*)srcB)[1];
      if (k0 + BK < DM) {                       // gfx1250 global_prefetch_b8
        __builtin_prefetch(srcA + BK, 0, 1);
        __builtin_prefetch(srcB + BK, 0, 1);
      }
      v8bf* dA = (v8bf*)(As + row * LDAB + seg * 16);
      v8bf* dB = (v8bf*)(Bs + row * LDAB + seg * 16);
      dA[0] = a0; dA[1] = a1;
      dB[0] = b0; dB[1] = b1;
    }
    __syncthreads();

    v16bf afr[2], bfr[4];
#pragma unroll
    for (int mt = 0; mt < 2; ++mt)
      afr[mt] = frag_load(As + (32 * wm + 16 * mt + l15) * LDAB, hl);
#pragma unroll
    for (int nt = 0; nt < 4; ++nt)
      bfr[nt] = frag_load(Bs + (64 * wn + 16 * nt + l15) * LDAB, hl);
#pragma unroll
    for (int mt = 0; mt < 2; ++mt)
#pragma unroll
      for (int nt = 0; nt < 4; ++nt)
        acc[mt][nt] = wmma_bf16(afr[mt], bfr[nt], acc[mt][nt]);
  }

  // epilogue: bias, convert to bf16, scatter to head-split layouts
#pragma unroll
  for (int mt = 0; mt < 2; ++mt) {
#pragma unroll
    for (int nt = 0; nt < 4; ++nt) {
      const int col = n0 + 64 * wn + 16 * nt + l15;   // n in [0,1024)
      const int head = col >> 6, d = col & 63;
      const float bval = bias[col];
#pragma unroll
      for (int r = 0; r < 8; ++r) {
        const int grow = m0 + 32 * wm + 16 * mt + 8 * hl + r;  // m in [0,8192)
        const int b = grow >> 11, t = grow & (SEQ - 1);
        const bf16 v = (bf16)(acc[mt][nt][r] + bval);
        if (proj == 2) {
          Vto[(((size_t)b * HEADS + head) * HD + d) * SEQ + t] = v;
        } else {
          bf16* dst = (proj == 0) ? Qo : Ko;
          dst[(((size_t)b * HEADS + head) * SEQ + t) * HD + d] = v;
        }
      }
    }
  }
}

// ---------------------------------------------------------------------------
// Flash attention (causal), bf16 WMMA, fp32 accum/softmax.
// Grid: (T/128, HEADS, BATCH). 256 threads = 8 waves; wave w owns q-rows
// [16w,16w+16) of the 128-row q-block. S and O live in registers; P is
// re-laid-out through LDS into the WMMA A-operand format.
// ---------------------------------------------------------------------------
#define QLD 72    // 64 + 8 pad halves (144B rows)
#define VLD 136   // 128 + 8 pad halves (272B rows)

__global__ __launch_bounds__(256)
void attn_fwd(const bf16* __restrict__ Q, const bf16* __restrict__ K,
              const bf16* __restrict__ Vt, bf16* __restrict__ O) {
  const int qb = blockIdx.x;            // q block, 0..15
  const int head = blockIdx.y;
  const int b = blockIdx.z;
  const int tid = threadIdx.x;
  const int lane = tid & 31, wave = tid >> 5;
  const int l15 = lane & 15, hl = lane >> 4;

  __shared__ __align__(16) bf16 Qs[128 * QLD];
  __shared__ __align__(16) bf16 Ks[128 * QLD];
  __shared__ __align__(16) bf16 Vts[64 * VLD];
  __shared__ __align__(16) bf16 Ps[128 * VLD];

  const size_t bh = (size_t)b * HEADS + head;
  const bf16* Qg  = Q  + (bh * SEQ + (size_t)qb * 128) * HD;
  const bf16* Kg  = K  + bh * SEQ * HD;
  const bf16* Vtg = Vt + bh * HD * SEQ;

  // load Q tile (128 x 64)
  {
    const int row = tid >> 1, seg = tid & 1;
    const v8bf* s = (const v8bf*)(Qg + row * HD + seg * 32);
    v8bf t0 = s[0], t1 = s[1], t2 = s[2], t3 = s[3];
    v8bf* d = (v8bf*)(Qs + row * QLD + seg * 32);
    d[0] = t0; d[1] = t1; d[2] = t2; d[3] = t3;
  }
  __syncthreads();
  const v16bf aq0 = frag_load(Qs + (16 * wave + l15) * QLD + 0, hl);
  const v16bf aq1 = frag_load(Qs + (16 * wave + l15) * QLD + 32, hl);

  float mrow[8], lrow[8];
  v8f o[4];
#pragma unroll
  for (int r = 0; r < 8; ++r) { mrow[r] = -1e30f; lrow[r] = 0.0f; }
#pragma unroll
  for (int t = 0; t < 4; ++t)
#pragma unroll
    for (int r = 0; r < 8; ++r) o[t][r] = 0.0f;

  const int qrow_base = qb * 128 + 16 * wave + 8 * hl;
  const float sm_scale = 0.125f;  // 1/sqrt(64)

  for (int j = 0; j <= qb; ++j) {
    __syncthreads();   // previous iteration done with Ks/Vts
    {
      const int row = tid >> 1, seg = tid & 1;
      const v8bf* s = (const v8bf*)(Kg + (size_t)(j * 128 + row) * HD + seg * 32);
      v8bf t0 = s[0], t1 = s[1], t2 = s[2], t3 = s[3];
      v8bf* d = (v8bf*)(Ks + row * QLD + seg * 32);
      d[0] = t0; d[1] = t1; d[2] = t2; d[3] = t3;
    }
    {
      const int dr = tid >> 2, seg = tid & 3;
      const v8bf* s = (const v8bf*)(Vtg + (size_t)dr * SEQ + j * 128 + seg * 32);
      v8bf t0 = s[0], t1 = s[1], t2 = s[2], t3 = s[3];
      v8bf* d = (v8bf*)(Vts + dr * VLD + seg * 32);
      d[0] = t0; d[1] = t1; d[2] = t2; d[3] = t3;
    }
    __syncthreads();

    // S = Q @ K^T : 8 n-tiles per wave, K-dim 64 = 2 WMMA steps
    v8f s[8];
#pragma unroll
    for (int t = 0; t < 8; ++t) {
      v16bf b0 = frag_load(Ks + (16 * t + l15) * QLD + 0, hl);
      v16bf b1 = frag_load(Ks + (16 * t + l15) * QLD + 32, hl);
      v8f c;
#pragma unroll
      for (int r = 0; r < 8; ++r) c[r] = 0.0f;
      c = wmma_bf16(aq0, b0, c);
      c = wmma_bf16(aq1, b1, c);
      s[t] = c;
    }

    // scale + causal mask
    const bool diag = (j == qb);
#pragma unroll
    for (int t = 0; t < 8; ++t) {
      const int kcol = j * 128 + 16 * t + l15;
#pragma unroll
      for (int r = 0; r < 8; ++r) {
        float v = s[t][r] * sm_scale;
        if (diag && kcol > qrow_base + r) v = -1e30f;
        s[t][r] = v;
      }
    }

    // online softmax: per-row (half-wave) reductions
    float mblk[8];
#pragma unroll
    for (int r = 0; r < 8; ++r) {
      float m = s[0][r];
#pragma unroll
      for (int t = 1; t < 8; ++t) m = fmaxf(m, s[t][r]);
      m = fmaxf(m, __shfl_xor(m, 1, 32));
      m = fmaxf(m, __shfl_xor(m, 2, 32));
      m = fmaxf(m, __shfl_xor(m, 4, 32));
      m = fmaxf(m, __shfl_xor(m, 8, 32));
      mblk[r] = m;
    }
    float alpha[8];
#pragma unroll
    for (int r = 0; r < 8; ++r) {
      const float mn = fmaxf(mrow[r], mblk[r]);
      alpha[r] = __expf(mrow[r] - mn);
      mrow[r] = mn;
    }
    float lblk[8];
#pragma unroll
    for (int r = 0; r < 8; ++r) lblk[r] = 0.0f;
#pragma unroll
    for (int t = 0; t < 8; ++t)
#pragma unroll
      for (int r = 0; r < 8; ++r) {
        const float p = __expf(s[t][r] - mrow[r]);
        s[t][r] = p;
        lblk[r] += p;
      }
#pragma unroll
    for (int r = 0; r < 8; ++r) {
      float l = lblk[r];
      l += __shfl_xor(l, 1, 32);
      l += __shfl_xor(l, 2, 32);
      l += __shfl_xor(l, 4, 32);
      l += __shfl_xor(l, 8, 32);
      lrow[r] = lrow[r] * alpha[r] + l;
    }
#pragma unroll
    for (int t = 0; t < 4; ++t)
#pragma unroll
      for (int r = 0; r < 8; ++r) o[t][r] *= alpha[r];

    // P -> LDS (per-wave stripe; C-layout -> A-layout re-stage)
#pragma unroll
    for (int t = 0; t < 8; ++t)
#pragma unroll
      for (int r = 0; r < 8; ++r)
        Ps[(16 * wave + 8 * hl + r) * VLD + 16 * t + l15] = (bf16)s[t][r];

    // O += P @ V : K-dim 128 = 4 WMMA steps, 4 n-tiles (d)
    v16bf ap[4];
#pragma unroll
    for (int kk = 0; kk < 4; ++kk)
      ap[kk] = frag_load(Ps + (16 * wave + l15) * VLD + 32 * kk, hl);
#pragma unroll
    for (int tn = 0; tn < 4; ++tn) {
      v8f c = o[tn];
#pragma unroll
      for (int kk = 0; kk < 4; ++kk) {
        v16bf bv = frag_load(Vts + (16 * tn + l15) * VLD + 32 * kk, hl);
        c = wmma_bf16(ap[kk], bv, c);
      }
      o[tn] = c;
    }
  }

  // finalize: O/l, write [b, t, head*64+d] bf16
#pragma unroll
  for (int tn = 0; tn < 4; ++tn)
#pragma unroll
    for (int r = 0; r < 8; ++r) {
      const int trow = qb * 128 + 16 * wave + 8 * hl + r;
      const float v = o[tn][r] / lrow[r];
      O[((size_t)b * SEQ + trow) * DM + head * HD + 16 * tn + l15] = (bf16)v;
    }
}

// ---------------------------------------------------------------------------
// Output projection: out = O @ Wo^T + bo, fp32 output.
// ---------------------------------------------------------------------------
__global__ __launch_bounds__(256)
void gemm_out(const bf16* __restrict__ A, const bf16* __restrict__ W,
              const float* __restrict__ bias, float* __restrict__ out) {
  const int m0 = blockIdx.y * 128;
  const int n0 = blockIdx.x * 128;
  const int tid = threadIdx.x;
  const int lane = tid & 31, wave = tid >> 5;
  const int l15 = lane & 15, hl = lane >> 4;
  const int wm = wave & 3, wn = wave >> 2;

  __shared__ __align__(16) bf16 As[128 * LDAB];
  __shared__ __align__(16) bf16 Bs[128 * LDAB];

  v8f acc[2][4];
#pragma unroll
  for (int mt = 0; mt < 2; ++mt)
#pragma unroll
    for (int nt = 0; nt < 4; ++nt)
#pragma unroll
      for (int r = 0; r < 8; ++r) acc[mt][nt][r] = 0.0f;

  const int row = tid >> 1, seg = tid & 1;

  for (int k0 = 0; k0 < DM; k0 += BK) {
    __syncthreads();
    {
      const bf16* srcA = A + (size_t)(m0 + row) * DM + k0 + seg * 16;
      const bf16* srcB = W + (size_t)(n0 + row) * DM + k0 + seg * 16;
      v8bf a0 = ((const v8bf*)srcA)[0], a1 = ((const v8bf*)srcA)[1];
      v8bf b0 = ((const v8bf*)srcB)[0], b1 = ((const v8bf*)srcB)[1];
      if (k0 + BK < DM) {
        __builtin_prefetch(srcA + BK, 0, 1);
        __builtin_prefetch(srcB + BK, 0, 1);
      }
      v8bf* dA = (v8bf*)(As + row * LDAB + seg * 16);
      v8bf* dB = (v8bf*)(Bs + row * LDAB + seg * 16);
      dA[0] = a0; dA[1] = a1;
      dB[0] = b0; dB[1] = b1;
    }
    __syncthreads();

    v16bf afr[2], bfr[4];
#pragma unroll
    for (int mt = 0; mt < 2; ++mt)
      afr[mt] = frag_load(As + (32 * wm + 16 * mt + l15) * LDAB, hl);
#pragma unroll
    for (int nt = 0; nt < 4; ++nt)
      bfr[nt] = frag_load(Bs + (64 * wn + 16 * nt + l15) * LDAB, hl);
#pragma unroll
    for (int mt = 0; mt < 2; ++mt)
#pragma unroll
      for (int nt = 0; nt < 4; ++nt)
        acc[mt][nt] = wmma_bf16(afr[mt], bfr[nt], acc[mt][nt]);
  }

#pragma unroll
  for (int mt = 0; mt < 2; ++mt)
#pragma unroll
    for (int nt = 0; nt < 4; ++nt) {
      const int col = n0 + 64 * wn + 16 * nt + l15;
      const float bval = bias[col];
#pragma unroll
      for (int r = 0; r < 8; ++r) {
        const int grow = m0 + 32 * wm + 16 * mt + 8 * hl + r;
        out[(size_t)grow * DM + col] = acc[mt][nt][r] + bval;
      }
    }
}

// ---------------------------------------------------------------------------
// Host launch
// ---------------------------------------------------------------------------
extern "C" void kernel_launch(void* const* d_in, const int* in_sizes, int n_in,
                              void* d_out, int out_size, void* d_ws, size_t ws_size,
                              hipStream_t stream) {
  (void)in_sizes; (void)n_in; (void)out_size; (void)ws_size;
  const float* x  = (const float*)d_in[0];
  const float* Wq = (const float*)d_in[1];
  const float* bq = (const float*)d_in[2];
  const float* Wk = (const float*)d_in[3];
  const float* bk = (const float*)d_in[4];
  const float* Wv = (const float*)d_in[5];
  const float* bv = (const float*)d_in[6];
  const float* Wo = (const float*)d_in[7];
  const float* bo = (const float*)d_in[8];

  const size_t NX = (size_t)ROWS * DM;   // 8388608
  const size_t NW = (size_t)DM * DM;     // 1048576

  bf16* xb  = (bf16*)d_ws;
  bf16* Wqb = xb + NX;
  bf16* Wkb = Wqb + NW;
  bf16* Wvb = Wkb + NW;
  bf16* Wob = Wvb + NW;
  bf16* Qb  = Wob + NW;
  bf16* Kb  = Qb + NX;
  bf16* Vtb = Kb + NX;
  bf16* Ob  = Vtb + NX;

  cvt_f32_bf16<<<(int)(NX / 1024), 256, 0, stream>>>(x,  xb,  (int)NX);
  cvt_f32_bf16<<<(int)(NW / 1024), 256, 0, stream>>>(Wq, Wqb, (int)NW);
  cvt_f32_bf16<<<(int)(NW / 1024), 256, 0, stream>>>(Wk, Wkb, (int)NW);
  cvt_f32_bf16<<<(int)(NW / 1024), 256, 0, stream>>>(Wv, Wvb, (int)NW);
  cvt_f32_bf16<<<(int)(NW / 1024), 256, 0, stream>>>(Wo, Wob, (int)NW);

  gemm_qkv<<<dim3(DM / 128, ROWS / 128, 3), 256, 0, stream>>>(
      xb, Wqb, Wkb, Wvb, bq, bk, bv, Qb, Kb, Vtb);

  attn_fwd<<<dim3(SEQ / 128, HEADS, BATCH), 256, 0, stream>>>(Qb, Kb, Vtb, Ob);

  gemm_out<<<dim3(DM / 128, ROWS / 128), 256, 0, stream>>>(Ob, Wob, bo,
                                                           (float*)d_out);
}
